// BertAttention_78554951843978
// MI455X (gfx1250) — matrix operand order repaired and checked by reading, not compiled, not run on hardware
//
#include <hip/hip_runtime.h>

// Problem sizes (fixed by reference)
#define BB  2
#define SS  2048
#define DD  1024
#define HH  16
#define HDQ 64
#define MM  (BB*SS)   // 4096 rows

typedef __attribute__((ext_vector_type(16))) __bf16        v16bf;
typedef __attribute__((ext_vector_type(8)))  float         v8f;
typedef __attribute__((ext_vector_type(8)))  unsigned int  v8u;
typedef __attribute__((__vector_size__(4 * sizeof(int)))) int v4i;
typedef __attribute__((address_space(1))) v4i gv4i;   // global 4xi32
typedef __attribute__((address_space(3))) v4i sv4i;   // LDS 4xi32

#if defined(__has_builtin)
#if __has_builtin(__builtin_amdgcn_global_load_async_to_lds_b128) && \
    __has_builtin(__builtin_amdgcn_s_wait_asynccnt)
#define USE_ASYNC_LDS 1
#endif
#endif

__device__ __forceinline__ unsigned short f2bf(float f) {
  unsigned int u = __builtin_bit_cast(unsigned int, f);
  u += 0x7FFFu + ((u >> 16) & 1u);          // round-to-nearest-even
  return (unsigned short)(u >> 16);
}
__device__ __forceinline__ unsigned int packbf(float lo, float hi) {
  return (unsigned int)f2bf(lo) | ((unsigned int)f2bf(hi) << 16);
}
__device__ __forceinline__ v8f wmma_bf16(v8u a, v8u b, v8f c) {
  return __builtin_amdgcn_wmma_f32_16x16x32_bf16(
      false, __builtin_bit_cast(v16bf, a),
      false, __builtin_bit_cast(v16bf, b),
      (short)0, c, false, false);
}
// A-matrix (16x32, 16-bit) per-lane K index for VGPR j (pair base)
__device__ __forceinline__ int akk(int j, int hv) { return ((j < 4) ? 2*j : 2*j + 8) + hv*8; }
// B-matrix (32x16, 16-bit) per-lane K index for VGPR j (pair base)
__device__ __forceinline__ int bkk(int j, int hv) { return ((j < 4) ? 2*j : 2*j + 8) + hv*8; }

// ---------------------------------------------------------------- convert
__global__ __launch_bounds__(256)
void cvt_bf16_k(const float* __restrict__ in, unsigned short* __restrict__ out, int n) {
  int i = blockIdx.x * 256 + threadIdx.x;
  if (i < n) out[i] = f2bf(in[i]);
}

// ---------------------------------------------------------------- fused QKV + proj GEMM
// grid (MM/32, DD/64, 4), block 32.  Y = X @ W^T (+bias); 32x64 tile per wave
__global__ __launch_bounds__(32)
void gemm_qkvp_k(const unsigned short* __restrict__ Xb,
                 const unsigned short* __restrict__ Wqb,
                 const unsigned short* __restrict__ Wkb,
                 const unsigned short* __restrict__ Wvb,
                 const unsigned short* __restrict__ Wpb,
                 const float* __restrict__ bq,
                 const float* __restrict__ bk,
                 const float* __restrict__ bvp,
                 unsigned short* __restrict__ Qo,
                 unsigned short* __restrict__ Ko,
                 unsigned short* __restrict__ Vo,
                 float* __restrict__ resO)
{
  const int lane = threadIdx.x;
  const int nl = lane & 15, hv = lane >> 4;
  const int m_base = blockIdx.x * 32;
  const int n_base = blockIdx.y * 64;
  const int mat = blockIdx.z;
  const unsigned short* W = (mat == 0) ? Wqb : (mat == 1) ? Wkb : (mat == 2) ? Wvb : Wpb;

  v8f acc[2][4];
  #pragma unroll
  for (int mi = 0; mi < 2; ++mi)
    #pragma unroll
    for (int c = 0; c < 4; ++c)
      #pragma unroll
      for (int r = 0; r < 8; ++r) acc[mi][c][r] = 0.f;

  const size_t mrow0 = (size_t)(m_base + nl) * DD;
  const size_t mrow1 = (size_t)(m_base + 16 + nl) * DD;
  for (int k0 = 0; k0 < DD; k0 += 32) {
    v8u a0, a1;
    #pragma unroll
    for (int j = 0; j < 8; ++j) {
      const int kk = akk(j, hv);
      a0[j] = *(const unsigned int*)(Xb + mrow0 + k0 + kk);
      a1[j] = *(const unsigned int*)(Xb + mrow1 + k0 + kk);
    }
    #pragma unroll
    for (int c = 0; c < 4; ++c) {
      const size_t wrow = (size_t)(n_base + c*16 + nl) * DD;
      v8u bm;
      #pragma unroll
      for (int j = 0; j < 8; ++j)
        bm[j] = *(const unsigned int*)(W + wrow + k0 + bkk(j, hv));
      acc[0][c] = wmma_bf16(a0, bm, acc[0][c]);   // B[k][n] = W[n][k]  (Y = X W^T)
      acc[1][c] = wmma_bf16(a1, bm, acc[1][c]);
    }
  }

  const int bidx = m_base / SS;            // whole 32-row tile shares batch index
  if (mat < 3) {
    const float* bias = (mat == 0) ? bq : (mat == 1) ? bk : bvp;
    unsigned short* out = (mat == 0) ? Qo : (mat == 1) ? Ko : Vo;
    #pragma unroll
    for (int c = 0; c < 4; ++c) {
      const int ncol = n_base + c*16 + nl;
      const float bval = bias[ncol];
      const int h = ncol >> 6, d = ncol & 63;
      #pragma unroll
      for (int mi = 0; mi < 2; ++mi)
        #pragma unroll
        for (int r = 0; r < 8; ++r) {
          const int s = (m_base % SS) + mi*16 + r + hv*8;
          out[(((size_t)bidx*HH + h)*SS + s)*HDQ + d] = f2bf(acc[mi][c][r] + bval);
        }
    }
  } else {
    #pragma unroll
    for (int c = 0; c < 4; ++c) {
      const int ncol = n_base + c*16 + nl;
      #pragma unroll
      for (int mi = 0; mi < 2; ++mi)
        #pragma unroll
        for (int r = 0; r < 8; ++r)
          resO[(size_t)(m_base + mi*16 + r + hv*8) * DD + ncol] = acc[mi][c][r];
    }
  }
}

// ---------------------------------------------------------------- flash attention
// grid (SS/16, HH, BB), block 32 (one wave per 16-row Q tile)
__global__ __launch_bounds__(32)
void attn_k(const unsigned short* __restrict__ Qb,
            const unsigned short* __restrict__ Kb,
            const unsigned short* __restrict__ Vb,
            float* __restrict__ ctx)
{
  __shared__ float P_lds[16 * 32];
  __shared__ __align__(16) unsigned short V_lds[32 * HDQ];  // row-major [key][d]

  const int lane = threadIdx.x, nl = lane & 15, hv = lane >> 4;
  const int qt = blockIdx.x, h = blockIdx.y, b = blockIdx.z;
  const size_t head = ((size_t)b * HH + h) * SS * HDQ;
  const unsigned short* Qp = Qb + head + (size_t)qt * 16 * HDQ;
  const unsigned short* Kp = Kb + head;
  const unsigned short* Vp = Vb + head;

  // Q tile as two A-matrices (d = 0..31, 32..63), loaded once
  v8u aq0, aq1;
  #pragma unroll
  for (int j = 0; j < 8; ++j) {
    const int kk = akk(j, hv);
    aq0[j] = *(const unsigned int*)(Qp + nl*HDQ + kk);
    aq1[j] = *(const unsigned int*)(Qp + nl*HDQ + 32 + kk);
  }

  float m_i[8], l_i[8];
  v8f o[4];
  #pragma unroll
  for (int r = 0; r < 8; ++r) { m_i[r] = -1e30f; l_i[r] = 0.f; }
  #pragma unroll
  for (int c = 0; c < 4; ++c)
    #pragma unroll
    for (int r = 0; r < 8; ++r) o[c][r] = 0.f;

  for (int kt = 0; kt < SS/16; kt += 2) {      // 32 keys per iteration
    // stage V tile (32 keys x 64) row-major into LDS via async copy
    {
      const unsigned short* vrow = Vp + (size_t)(kt*16 + lane) * HDQ;
      unsigned short* lrow = &V_lds[lane * HDQ];
#if defined(USE_ASYNC_LDS)
      #pragma unroll
      for (int i = 0; i < 8; ++i)
        __builtin_amdgcn_global_load_async_to_lds_b128(
            (gv4i*)(vrow + i*8), (sv4i*)(lrow + i*8), 0, 0);
      __builtin_amdgcn_s_wait_asynccnt(0);
#else
      #pragma unroll
      for (int i = 0; i < 8; ++i)
        *(uint4*)(lrow + i*8) = *(const uint4*)(vrow + i*8);
#endif
    }
    // prefetch next K/V tiles while we do the matrix + softmax work
    if (kt + 2 < SS/16) {
      __builtin_prefetch(Kp + (size_t)((kt+2)*16 + lane) * HDQ, 0, 0);
      __builtin_prefetch(Vp + (size_t)((kt+2)*16 + lane) * HDQ, 0, 0);
    }
    // scores: two 16x16 tiles, each chained over d (K=32 twice)
    v8f sc[2];
    #pragma unroll
    for (int g = 0; g < 2; ++g) {
      #pragma unroll
      for (int r = 0; r < 8; ++r) sc[g][r] = 0.f;
      const unsigned short* krow = Kp + (size_t)((kt + g)*16 + nl) * HDQ;
      v8u b0, b1;
      #pragma unroll
      for (int j = 0; j < 8; ++j) {
        const int kk = bkk(j, hv);
        b0[j] = *(const unsigned int*)(krow + kk);
        b1[j] = *(const unsigned int*)(krow + 32 + kk);
      }
      sc[g] = wmma_bf16(aq0, b0, sc[g]);
      sc[g] = wmma_bf16(aq1, b1, sc[g]);
    }
    // online softmax (rows = r + 8*hv; cols across the 16-lane half)
    float mnew[8], corr[8];
    #pragma unroll
    for (int r = 0; r < 8; ++r) {
      const float s0 = sc[0][r] * 0.125f;       // 1/sqrt(64)
      const float s1 = sc[1][r] * 0.125f;
      sc[0][r] = s0; sc[1][r] = s1;
      float mx = fmaxf(s0, s1);
      #pragma unroll
      for (int d = 1; d < 16; d <<= 1) mx = fmaxf(mx, __shfl_xor(mx, d, 32));
      mnew[r] = fmaxf(m_i[r], mx);
      corr[r] = __expf(m_i[r] - mnew[r]);
      m_i[r]  = mnew[r];
    }
    #pragma unroll
    for (int r = 0; r < 8; ++r) {
      const float p0 = __expf(sc[0][r] - mnew[r]);
      const float p1 = __expf(sc[1][r] - mnew[r]);
      const int row = r + 8*hv;
      P_lds[row*32 + nl]      = p0;
      P_lds[row*32 + 16 + nl] = p1;
      float rs = p0 + p1;
      #pragma unroll
      for (int d = 1; d < 16; d <<= 1) rs += __shfl_xor(rs, d, 32);
      l_i[r] = l_i[r] * corr[r] + rs;
    }
    __syncthreads();
    // P (16x32) as A-matrix via LDS transpose
    v8u pa;
    #pragma unroll
    for (int j = 0; j < 8; ++j) {
      const int kk = akk(j, hv);
      pa[j] = packbf(P_lds[nl*32 + kk], P_lds[nl*32 + kk + 1]);
    }
    // o += P @ V   (B[k][d] read transposed from row-major V_lds)
    #pragma unroll
    for (int c = 0; c < 4; ++c) {
      #pragma unroll
      for (int r = 0; r < 8; ++r) o[c][r] *= corr[r];
      const int d = c*16 + nl;
      v8u vb;
      #pragma unroll
      for (int j = 0; j < 8; ++j) {
        const int k = bkk(j, hv);
        vb[j] = (unsigned int)V_lds[k*HDQ + d] |
                ((unsigned int)V_lds[(k+1)*HDQ + d] << 16);
      }
      o[c] = wmma_bf16(pa, vb, o[c]);
    }
    __syncthreads();
  }
  // finalize: ctx[b, s, h*64 + d]
  #pragma unroll
  for (int c = 0; c < 4; ++c) {
    const int dcol = h*HDQ + c*16 + nl;
    #pragma unroll
    for (int r = 0; r < 8; ++r) {
      const int m = b*SS + qt*16 + r + 8*hv;
      ctx[(size_t)m*DD + dcol] = o[c][r] / l_i[r];
    }
  }
}

// ---------------------------------------------------------------- relu + residual + layernorm
// grid MM, block 256 (one row each)
__global__ __launch_bounds__(256)
void relu_ln_k(const float* __restrict__ ctx, const float* __restrict__ res,
               const float* __restrict__ gamma, const float* __restrict__ beta,
               float* __restrict__ out)
{
  __shared__ float ls[8], ls2[8];
  const int row = blockIdx.x, t = threadIdx.x;
  float x[4], s = 0.f, s2 = 0.f;
  #pragma unroll
  for (int i = 0; i < 4; ++i) {
    const int idx = t + 256*i;
    float v = ctx[(size_t)row*DD + idx] + res[(size_t)row*DD + idx];
    v = fmaxf(v, 0.f);
    x[i] = v; s += v; s2 += v*v;
  }
  #pragma unroll
  for (int d = 1; d < 32; d <<= 1) { s += __shfl_xor(s, d, 32); s2 += __shfl_xor(s2, d, 32); }
  if ((t & 31) == 0) { ls[t >> 5] = s; ls2[t >> 5] = s2; }
  __syncthreads();
  float ts = 0.f, ts2 = 0.f;
  #pragma unroll
  for (int i = 0; i < 8; ++i) { ts += ls[i]; ts2 += ls2[i]; }
  const float mu  = ts * (1.f / DD);
  const float var = ts2 * (1.f / DD) - mu*mu;
  const float inv = rsqrtf(var + 1e-12f);
  #pragma unroll
  for (int i = 0; i < 4; ++i) {
    const int idx = t + 256*i;
    out[(size_t)row*DD + idx] = (x[i] - mu) * inv * gamma[idx] + beta[idx];
  }
}

// ---------------------------------------------------------------- host
extern "C" void kernel_launch(void* const* d_in, const int* in_sizes, int n_in,
                              void* d_out, int out_size, void* d_ws, size_t ws_size,
                              hipStream_t stream) {
  const float* X   = (const float*)d_in[0];
  const float* Wq  = (const float*)d_in[1];
  const float* bq  = (const float*)d_in[2];
  const float* Wk  = (const float*)d_in[3];
  const float* bk  = (const float*)d_in[4];
  const float* Wv  = (const float*)d_in[5];
  const float* bvp = (const float*)d_in[6];
  const float* Wp  = (const float*)d_in[7];
  const float* gamma = (const float*)d_in[8];
  const float* beta  = (const float*)d_in[9];
  float* out = (float*)d_out;

  char* base = (char*)d_ws;
  size_t off = 0;
  auto carve = [&](size_t bytes) { char* p = base + off; off = (off + bytes + 255) & ~(size_t)255; return p; };
  unsigned short* Xb  = (unsigned short*)carve((size_t)MM*DD*2);
  unsigned short* Wqb = (unsigned short*)carve((size_t)DD*DD*2);
  unsigned short* Wkb = (unsigned short*)carve((size_t)DD*DD*2);
  unsigned short* Wvb = (unsigned short*)carve((size_t)DD*DD*2);
  unsigned short* Wpb = (unsigned short*)carve((size_t)DD*DD*2);
  unsigned short* Qb  = (unsigned short*)carve((size_t)MM*DD*2);
  unsigned short* Kb  = (unsigned short*)carve((size_t)MM*DD*2);
  unsigned short* Vb  = (unsigned short*)carve((size_t)MM*DD*2);
  float* resb = (float*)carve((size_t)MM*DD*4);
  float* ctxb = (float*)carve((size_t)MM*DD*4);

  const int nX = MM*DD, nW = DD*DD;
  cvt_bf16_k<<<(nX + 255)/256, 256, 0, stream>>>(X,  Xb,  nX);
  cvt_bf16_k<<<(nW + 255)/256, 256, 0, stream>>>(Wq, Wqb, nW);
  cvt_bf16_k<<<(nW + 255)/256, 256, 0, stream>>>(Wk, Wkb, nW);
  cvt_bf16_k<<<(nW + 255)/256, 256, 0, stream>>>(Wv, Wvb, nW);
  cvt_bf16_k<<<(nW + 255)/256, 256, 0, stream>>>(Wp, Wpb, nW);

  gemm_qkvp_k<<<dim3(MM/32, DD/64, 4), 32, 0, stream>>>(
      Xb, Wqb, Wkb, Wvb, Wpb, bq, bk, bvp, Qb, Kb, Vb, resb);

  attn_k<<<dim3(SS/16, HH, BB), 32, 0, stream>>>(Qb, Kb, Vb, ctxb);

  relu_ln_k<<<MM, 256, 0, stream>>>(ctxb, resb, gamma, beta, out);
}
